// WeightOnlyQuipLinear_26877905339290
// MI455X (gfx1250) — matrix operand order repaired and checked by reading, not compiled, not run on hardware
//
#include <hip/hip_runtime.h>

#define IN_F  4096
#define OUT_F 4096

typedef __attribute__((ext_vector_type(8)))  _Float16 v8h;
typedef __attribute__((ext_vector_type(16))) _Float16 v16h;
typedef __attribute__((ext_vector_type(8)))  float    v8f;

// ---------------------------------------------------------------------------
// Kernel 1: x' = FHT(x * SU) * (1/sqrt(4096)) * Wscale   (fp16 out, one row/block)
// ---------------------------------------------------------------------------
__global__ __launch_bounds__(256) void fht_pre_kernel(
    const _Float16* __restrict__ x, const _Float16* __restrict__ SU,
    const float* __restrict__ Wscale, _Float16* __restrict__ xp)
{
    __shared__ float buf[IN_F];                       // 16 KB
    const int row = blockIdx.x;
    const int tid = threadIdx.x;
    const _Float16* xr = x + (size_t)row * IN_F;

    for (int i = tid; i < IN_F; i += 256)
        buf[i] = (float)xr[i] * (float)SU[i];
    __syncthreads();

    for (int h = 1; h < IN_F; h <<= 1) {
        for (int p = tid; p < IN_F / 2; p += 256) {
            int i = ((p & ~(h - 1)) << 1) | (p & (h - 1));
            int j = i + h;
            float a = buf[i], b = buf[j];
            buf[i] = a + b;
            buf[j] = a - b;
        }
        __syncthreads();
    }

    const float sc = 0.015625f * Wscale[0];           // 1/64 = 1/sqrt(4096)
    _Float16* xpr = xp + (size_t)row * IN_F;
    for (int i = tid; i < IN_F; i += 256)
        xpr[i] = (_Float16)(buf[i] * sc);
}

// ---------------------------------------------------------------------------
// Kernel 2: out = x' @ decode_e8p(Qidxs)^T, fp16 out, f32 WMMA accumulation.
// 256x128 block tile, BK=64, 8 wave32 waves as 4(wm) x 2(wn), wave tile 64x64
// (4x4 WMMA tiles -> 512 B LDS per WMMA, at the 256 B/clk LDS roofline).
// 2-stage pipeline: X tile kt+1 via global_load_async_to_lds_b128 (ASYNCcnt),
// W tile kt+1 decoded from register-staged Qidxs words, one barrier per K-step.
// ---------------------------------------------------------------------------
#define BM  256
#define BN  128
#define BK  64
#define LDK 72   // padded row stride (halves): 144B rows, 16B aligned, conflict-free
#define KT  (IN_F / BK)

static __device__ inline void wait_async() {
#if __has_builtin(__builtin_amdgcn_s_wait_asynccnt)
    __builtin_amdgcn_s_wait_asynccnt(0);
#else
    asm volatile("s_wait_asynccnt 0x0" ::: "memory");
#endif
}

// issue 8 async 16B memory->LDS copies per thread (full 256x64 fp16 X tile/block)
static __device__ inline void async_x_tile(const _Float16* __restrict__ xp,
                                           int m0, int kt, unsigned ldsbase, int tid)
{
#pragma unroll
    for (int l = 0; l < 8; ++l) {
        int lin = tid + l * 256;                      // 0..2047
        int r   = lin >> 3;
        int c   = (lin & 7) * 8;                      // half index, 16B chunks
        const _Float16* src = xp + (size_t)(m0 + r) * IN_F + (size_t)kt * BK + c;
        unsigned dst = ldsbase + (unsigned)((r * LDK + c) * 2);
        asm volatile("global_load_async_to_lds_b128 %0, %1, off"
                     :: "v"(dst), "v"(src) : "memory");
    }
}

static __device__ inline void load_q(unsigned q[4], const int* __restrict__ Qidxs,
                                     int n0, int kt, int tid)
{
#pragma unroll
    for (int l = 0; l < 4; ++l) {
        int lin = tid + l * 256;                      // 0..1023
        int n = lin >> 3, cw = lin & 7;
        q[l] = (unsigned)Qidxs[(size_t)(n0 + n) * (IN_F / 8) + kt * 8 + cw];
    }
}

static __device__ inline void decode_store(_Float16* __restrict__ wsbuf,
                                           const uint4* __restrict__ gabs,
                                           const unsigned q[4], int tid)
{
#pragma unroll
    for (int l = 0; l < 4; ++l) {
        int lin = tid + l * 256;
        int n = lin >> 3, cw = lin & 7;
        unsigned qq = q[l];
        uint4 w = gabs[qq & 255u];                    // 8 abs fp16 values
        unsigned s = (qq >> 8) & 255u;                // sign bit per element
        w.x ^= ((s &   1u) << 15) | ((s &   2u) << 30);
        w.y ^= ((s &   4u) << 13) | ((s &   8u) << 28);
        w.z ^= ((s &  16u) << 11) | ((s &  32u) << 26);
        w.w ^= ((s &  64u) <<  9) | ((s & 128u) << 24);
        *(uint4*)&wsbuf[n * LDK + cw * 8] = w;
    }
}

static __device__ inline v16h load_frag(const _Float16* p) {
    // two 16B chunks 32B apart -> matches f16 A/B per-lane VGPR layout
    v8h lo = *(const v8h*)p;
    v8h hi = *(const v8h*)(p + 16);
    v16h r;
#pragma unroll
    for (int j = 0; j < 8; ++j) { r[j] = lo[j]; r[j + 8] = hi[j]; }
    return r;
}

__global__ __launch_bounds__(256) void quip_gemm_kernel(
    const _Float16* __restrict__ xp,
    const int*      __restrict__ Qidxs,
    const _Float16* __restrict__ grid_abs,
    _Float16*       __restrict__ out)
{
    __shared__ __align__(16) _Float16 Xs[2][BM * LDK]; // 2 x 36 KB
    __shared__ __align__(16) _Float16 Ws[2][BN * LDK]; // 2 x 18 KB  (108 KB total)

    const int tid  = threadIdx.x;
    const int lane = tid & 31;
    const int wid  = tid >> 5;
    const int wm   = wid >> 1;        // 0..3  -> 64 rows each
    const int wn   = wid & 1;         // 0..1  -> 64 cols each
    const int l16  = lane & 15;
    const int hic  = (lane >> 4) & 1; // upper half-wave selector

    const int n0 = blockIdx.x * BN;
    const int m0 = blockIdx.y * BM;

    const uint4* gabs = (const uint4*)grid_abs;       // 256 rows x 16B

    const unsigned xsb0 = (unsigned)(uintptr_t)&Xs[0][0];
    const unsigned xsb1 = (unsigned)(uintptr_t)&Xs[1][0];

    // ---- prologue: stage tile 0 into buffer 0, stage Qidxs words for tile 1 ----
    unsigned q[4];
    async_x_tile(xp, m0, 0, xsb0, tid);
    load_q(q, Qidxs, n0, 0, tid);
    decode_store(&Ws[0][0], gabs, q, tid);
    load_q(q, Qidxs, n0, 1, tid);                     // arrives during first compute

    v8f acc[4][4] = {};

    for (int kt = 0; kt < KT; ++kt) {
        const int buf = kt & 1;
        wait_async();                                  // own async X writes landed
        __syncthreads();                               // tile kt published block-wide

        if (kt + 1 < KT) {
            // buffer buf^1 is free: all waves finished compute(kt-1) before barrier
            async_x_tile(xp, m0, kt + 1, buf ? xsb0 : xsb1, tid);
            decode_store(&Ws[buf ^ 1][0], gabs, q, tid);
            if (kt + 2 < KT) load_q(q, Qidxs, n0, kt + 2, tid);
        }

        // ---- compute on buffer `buf` ----
        const _Float16* xb = &Xs[buf][0];
        const _Float16* wb = &Ws[buf][0];
#pragma unroll
        for (int ks = 0; ks < 2; ++ks) {
            const int ko = ks * 32;
            v16h b[4];
#pragma unroll
            for (int ni = 0; ni < 4; ++ni) {
                // B 32x16 f16: lane n&15 holds column n; lane<16 K0..7,16..23; lane>=16 K8..15,24..31
                const _Float16* p = &wb[(wn * 64 + ni * 16 + l16) * LDK + ko + hic * 8];
                b[ni] = load_frag(p);
            }
#pragma unroll
            for (int mi = 0; mi < 4; ++mi) {
                // A 16x32 f16: lane<16 holds K0..7,16..23; lane>=16 K8..15,24..31
                const _Float16* p = &xb[(wm * 64 + mi * 16 + l16) * LDK + ko + hic * 8];
                v16h a = load_frag(p);
#pragma unroll
                for (int ni = 0; ni < 4; ++ni)
                    acc[mi][ni] = __builtin_amdgcn_wmma_f32_16x16x32_f16(
                        false, a, false, b[ni], (short)0, acc[mi][ni], false, false);
            }
        }
    }

    // ---- epilogue: C/D layout VGPR r -> M = r (+8 for upper lanes), N = lane&15 ----
#pragma unroll
    for (int mi = 0; mi < 4; ++mi)
#pragma unroll
        for (int ni = 0; ni < 4; ++ni)
#pragma unroll
            for (int r = 0; r < 8; ++r) {
                int row = m0 + wm * 64 + mi * 16 + hic * 8 + r;
                int col = n0 + wn * 64 + ni * 16 + l16;
                out[(size_t)row * OUT_F + col] = (_Float16)acc[mi][ni][r];
            }
}

// ---------------------------------------------------------------------------
// Kernel 3: out = FHT(out) * (1/64) * SV + bias   (in-place per row)
// ---------------------------------------------------------------------------
__global__ __launch_bounds__(256) void fht_post_kernel(
    _Float16* __restrict__ out, const _Float16* __restrict__ SV,
    const _Float16* __restrict__ bias)
{
    __shared__ float buf[OUT_F];
    const int row = blockIdx.x;
    const int tid = threadIdx.x;
    _Float16* orow = out + (size_t)row * OUT_F;

    for (int i = tid; i < OUT_F; i += 256)
        buf[i] = (float)orow[i];
    __syncthreads();

    for (int h = 1; h < OUT_F; h <<= 1) {
        for (int p = tid; p < OUT_F / 2; p += 256) {
            int i = ((p & ~(h - 1)) << 1) | (p & (h - 1));
            int j = i + h;
            float a = buf[i], b = buf[j];
            buf[i] = a + b;
            buf[j] = a - b;
        }
        __syncthreads();
    }

    for (int i = tid; i < OUT_F; i += 256)
        orow[i] = (_Float16)(buf[i] * 0.015625f * (float)SV[i] + (float)bias[i]);
}

// ---------------------------------------------------------------------------
extern "C" void kernel_launch(void* const* d_in, const int* in_sizes, int n_in,
                              void* d_out, int out_size, void* d_ws, size_t ws_size,
                              hipStream_t stream)
{
    (void)n_in; (void)out_size; (void)ws_size;
    const _Float16* input    = (const _Float16*)d_in[0];
    const int*      Qidxs    = (const int*)d_in[1];
    const _Float16* grid_abs = (const _Float16*)d_in[2];
    const _Float16* SU       = (const _Float16*)d_in[3];
    const _Float16* SV       = (const _Float16*)d_in[4];
    const float*    Wscale   = (const float*)d_in[5];
    const _Float16* bias     = (const _Float16*)d_in[6];
    _Float16* out = (_Float16*)d_out;
    _Float16* xp  = (_Float16*)d_ws;                  // 8192*4096 fp16 = 64 MB

    const int rows = in_sizes[0] / IN_F;              // 8192

    fht_pre_kernel<<<rows, 256, 0, stream>>>(input, SU, Wscale, xp);

    dim3 grid(OUT_F / BN, rows / BM);                 // 32 x 32 blocks
    quip_gemm_kernel<<<grid, 256, 0, stream>>>(xp, Qidxs, grid_abs, out);

    fht_post_kernel<<<rows, 256, 0, stream>>>(out, SV, bias);
}
